// BidirectionalAttention_90769838833821
// MI455X (gfx1250) — matrix operand order repaired and checked by reading, not compiled, not run on hardware
//
#include <hip/hip_runtime.h>
#include <hip/hip_bf16.h>

// ---------------------------------------------------------------------------
// BidirectionalAttention on MI455X (gfx1250), wave32 + bf16 WMMA.
//   x:[B,C,H,W] f32, B=4,C=256,H=W=64 -> N=4096, C8=32
//   q,k = Wq/Wk @ x (C8 out), v = Wv @ x, attn = softmax(q^T k), out = v@attn^T
//   y = x + gamma*out ; returns (y,y) concatenated in d_out.
// v2: flash attention with workgroup-shared K/V chunks double-buffered in LDS
//     via CDNA5 async-to-LDS loads (ASYNCcnt) -> 4x less L2 traffic on V.
// ---------------------------------------------------------------------------

typedef __bf16 bf16_t;
typedef __attribute__((ext_vector_type(16))) __bf16 v16bf;
typedef __attribute__((ext_vector_type(8)))  __bf16 v8bf;
typedef __attribute__((ext_vector_type(8)))  float  v8f;
typedef int v4i __attribute__((vector_size(16)));
typedef __attribute__((address_space(1))) v4i* gv4i_p;   // global v4i*
typedef __attribute__((address_space(3))) v4i* lv4i_p;   // LDS v4i*

union FragAB { v16bf v; v8bf h[2]; };

#define B_  4
#define C_  256
#define C8_ 32
#define N_  4096
#define KVSTR 40   // padded LDS row stride (halves) for K/V chunks

__device__ __forceinline__ v8f vzero8() {
  v8f z = {0.f,0.f,0.f,0.f,0.f,0.f,0.f,0.f};
  return z;
}

__device__ __forceinline__ v8f wmma_bf16(v16bf a, v16bf b, v8f c) {
  return __builtin_amdgcn_wmma_f32_16x16x32_bf16(false, a, false, b,
                                                 (short)0, c, false, false);
}

// ---- CDNA5 async global->LDS copy (16B per lane), ASYNCcnt-tracked ---------
__device__ __forceinline__ void async_b128(const bf16_t* g, bf16_t* l) {
#if __has_builtin(__builtin_amdgcn_global_load_async_to_lds_b128)
  __builtin_amdgcn_global_load_async_to_lds_b128(
      (gv4i_p)(unsigned long long)(const void*)g,
      (lv4i_p)(unsigned)(unsigned long long)(void*)l,
      0, 0);
#else
  unsigned lds = (unsigned)(unsigned long long)(void*)l;
  asm volatile("global_load_async_to_lds_b128 %0, %1, off"
               :: "v"(lds), "v"((unsigned long long)g) : "memory");
#endif
}

__device__ __forceinline__ void wait_async0() {
#if __has_builtin(__builtin_amdgcn_s_wait_asynccnt)
  __builtin_amdgcn_s_wait_asynccnt(0);
#else
  asm volatile("s_wait_asynccnt 0x0" ::: "memory");
#endif
}

// ---- Pass 0a: x [B,C,N] f32 -> Xt [B,N,C] bf16 (transpose + convert) -------
__global__ void convert_x_kernel(const float* __restrict__ x,
                                 bf16_t* __restrict__ Xt) {
  int i = blockIdx.x * 256 + threadIdx.x;       // B*C*N threads
  int n  = i & (N_ - 1);
  int bc = i >> 12;
  int c  = bc & (C_ - 1);
  int b  = bc >> 8;
  Xt[((size_t)(b * N_ + n)) * C_ + c] = (bf16_t)x[i];
}

// ---- Pass 0b: weights f32 -> bf16 ------------------------------------------
__global__ void convert_w_kernel(const float* __restrict__ Wq,
                                 const float* __restrict__ Wk,
                                 const float* __restrict__ Wv,
                                 bf16_t* __restrict__ Wqb,
                                 bf16_t* __restrict__ Wkb,
                                 bf16_t* __restrict__ Wvb) {
  int i = blockIdx.x * 256 + threadIdx.x;       // 81920
  if (i < 8192)        Wqb[i]         = (bf16_t)Wq[i];
  else if (i < 16384)  Wkb[i - 8192]  = (bf16_t)Wk[i - 8192];
  else                 Wvb[i - 16384] = (bf16_t)Wv[i - 16384];
}

// ---- Pass 1: projections q,k,v via WMMA ------------------------------------
__global__ void proj_kernel(const bf16_t* __restrict__ Xt,
                            const bf16_t* __restrict__ Wqb,
                            const bf16_t* __restrict__ Wkb,
                            const bf16_t* __restrict__ Wvb,
                            const float* __restrict__ bq,
                            const float* __restrict__ bk,
                            const float* __restrict__ bv,
                            bf16_t* __restrict__ Qb,   // [B][N][32]
                            bf16_t* __restrict__ Kb,   // [B][N][32]
                            bf16_t* __restrict__ Vt) { // [B][C][N]
  int wave = (blockIdx.x * blockDim.x + threadIdx.x) >> 5; // 20480 waves
  int lane = threadIdx.x & 31;
  int t    = wave % 20;
  int rest = wave / 20;
  int nt   = rest & 255;
  int b    = rest >> 8;
  int n0   = nt * 16;

  const bf16_t* Wsel; const float* bsel; int o0, mode;
  if (t < 2)      { Wsel = Wqb; bsel = bq; o0 = t * 16;        mode = 0; }
  else if (t < 4) { Wsel = Wkb; bsel = bk; o0 = (t - 2) * 16;  mode = 1; }
  else            { Wsel = Wvb; bsel = bv; o0 = (t - 4) * 16;  mode = 2; }

  int rA = lane & 15;
  int hs = lane >> 4;

  const bf16_t* abase = Xt + ((size_t)(b * N_ + n0)) * C_;
  const bf16_t* wbase = Wsel + (size_t)(o0 + rA) * C_;

  v8f acc = vzero8();
#pragma unroll
  for (int k0 = 0; k0 < C_; k0 += 32) {
    FragAB A, Bv;
    const bf16_t* ap = abase + rA * C_ + k0 + hs * 8;
    A.h[0] = *(const v8bf*)(ap);
    A.h[1] = *(const v8bf*)(ap + 16);
    const bf16_t* wp = wbase + k0 + hs * 8;
    Bv.h[0] = *(const v8bf*)(wp);
    Bv.h[1] = *(const v8bf*)(wp + 16);
    acc = wmma_bf16(A.v, Bv.v, acc);
  }

  float bias = bsel[o0 + rA];
#pragma unroll
  for (int r = 0; r < 8; r++) {
    float val = acc[r] + bias;
    int n = n0 + r + hs * 8;
    if (mode == 0)      Qb[(size_t)(b * N_ + n) * C8_ + o0 + rA] = (bf16_t)val;
    else if (mode == 1) Kb[(size_t)(b * N_ + n) * C8_ + o0 + rA] = (bf16_t)val;
    else                Vt[(size_t)(b * C_ + o0 + rA) * N_ + n]  = (bf16_t)val;
  }
}

// ---- Pass 2: flash attention, WG-shared double-buffered K/V in LDS ---------
// Workgroup = 4 waves = 64 queries. Per 32-key chunk: async-stage K(32x32)
// and V(256x32) into LDS, each wave: 2 WMMA (S), online softmax, 16 WMMA (PV).
__global__ void __launch_bounds__(128)
attn_kernel(const bf16_t* __restrict__ Qb,
            const bf16_t* __restrict__ Kb,
            const bf16_t* __restrict__ Vt,
            const float* __restrict__ x,
            const float* __restrict__ gamma,
            float* __restrict__ out) {
  __shared__ __align__(16) bf16_t Kl[2][32 * KVSTR];    //  2 x 2.5 KB
  __shared__ __align__(16) bf16_t Vl[2][256 * KVSTR];   //  2 x 20  KB
  __shared__ __align__(16) bf16_t Pl[4][16 * 32];       //  4 x 1   KB

  int tid  = threadIdx.x;
  int wv   = tid >> 5;
  int lane = tid & 31;
  int b    = blockIdx.x >> 6;        // 256 blocks = B * (N/64)
  int qblk = blockIdx.x & 63;
  int n0   = qblk * 64 + wv * 16;

  int rA = lane & 15;
  int hs = lane >> 4;

  const bf16_t* kbase = Kb + (size_t)(b * N_) * C8_;
  const bf16_t* vbase = Vt + (size_t)(b * C_) * N_;

  // Stage one 32-key chunk (K: 32x32, V: 256x32 bf16) into LDS buffer `buf`.
  // 1152 16B transfers, 9 per thread, branch uniform per unrolled step.
  auto stage = [&](int m0, int buf) {
#pragma unroll
    for (int it = 0; it < 9; it++) {
      int j = tid + it * 128;
      if (it < 8) {                     // V: 1024 transfers
        int row = j >> 2, seg = j & 3;
        async_b128(vbase + (size_t)row * N_ + m0 + seg * 8,
                   &Vl[buf][row * KVSTR + seg * 8]);
      } else {                          // K: 128 transfers
        int r2 = j - 1024, row = r2 >> 2, seg = r2 & 3;
        async_b128(kbase + (size_t)(m0 + row) * C8_ + seg * 8,
                   &Kl[buf][row * KVSTR + seg * 8]);
      }
    }
  };

  // Q fragment (A operand, 16n x 32c), held in VGPRs for the whole loop.
  FragAB QA;
  {
    const bf16_t* qp = Qb + (size_t)(b * N_ + n0) * C8_ + rA * C8_ + hs * 8;
    QA.h[0] = *(const v8bf*)(qp);
    QA.h[1] = *(const v8bf*)(qp + 16);
  }

  float mrow[8], lrow[8], alph[8];
  v8f O[16];
#pragma unroll
  for (int r = 0; r < 8; r++) { mrow[r] = -__builtin_inff(); lrow[r] = 0.f; }
#pragma unroll
  for (int tt = 0; tt < 16; tt++) O[tt] = vzero8();

  stage(0, 0);                          // prologue: chunk 0 in flight

  for (int ch = 0; ch < N_ / 32; ch++) {
    wait_async0();                      // my chunk-ch transfers complete
    __syncthreads();                    // everyone's complete; prior compute done
    if (ch + 1 < N_ / 32) stage((ch + 1) * 32, (ch + 1) & 1);  // overlap next
    int buf = ch & 1;

    // --- S = Q K^T for the 32 staged keys (two 16-wide tiles) ---
    FragAB B0, B1;
    {
      const bf16_t* k0p = &Kl[buf][rA * KVSTR + hs * 8];
      const bf16_t* k1p = &Kl[buf][(16 + rA) * KVSTR + hs * 8];
      B0.h[0] = *(const v8bf*)(k0p);  B0.h[1] = *(const v8bf*)(k0p + 16);
      B1.h[0] = *(const v8bf*)(k1p);  B1.h[1] = *(const v8bf*)(k1p + 16);
    }
    v8f S0 = wmma_bf16(QA.v, B0.v, vzero8());
    v8f S1 = wmma_bf16(QA.v, B1.v, vzero8());

    // --- online softmax per row ---
#pragma unroll
    for (int r = 0; r < 8; r++) {
      float s0 = S0[r], s1 = S1[r];
      float mx = fmaxf(s0, s1);
#pragma unroll
      for (int off = 1; off < 16; off <<= 1)
        mx = fmaxf(mx, __shfl_xor(mx, off, 16));
      float mnew = fmaxf(mrow[r], mx);
      float a    = __expf(mrow[r] - mnew);
      float p0   = __expf(s0 - mnew);
      float p1   = __expf(s1 - mnew);
      float rs   = p0 + p1;
#pragma unroll
      for (int off = 1; off < 16; off <<= 1)
        rs += __shfl_xor(rs, off, 16);
      lrow[r] = lrow[r] * a + rs;
      mrow[r] = mnew;
      alph[r] = a;
      int row = r + hs * 8;                        // C/D layout row
      Pl[wv][row * 32 + rA]      = (bf16_t)p0;     // cols m..m+15
      Pl[wv][row * 32 + 16 + rA] = (bf16_t)p1;     // cols m+16..m+31
    }

#pragma unroll
    for (int tt = 0; tt < 16; tt++)
#pragma unroll
      for (int r = 0; r < 8; r++)
        O[tt][r] *= alph[r];

    // wave-local ordering: P stores land before fragment reload
    asm volatile("s_wait_dscnt 0" ::: "memory");

    FragAB PA;
    {
      const bf16_t* pp = &Pl[wv][rA * 32 + hs * 8];
      PA.h[0] = *(const v8bf*)(pp);
      PA.h[1] = *(const v8bf*)(pp + 16);
    }

    // --- O += P * V from staged LDS V, 16 channel tiles ---
#pragma unroll
    for (int tt = 0; tt < 16; tt++) {
      FragAB VB;
      const bf16_t* vp = &Vl[buf][(tt * 16 + rA) * KVSTR + hs * 8];
      VB.h[0] = *(const v8bf*)(vp);
      VB.h[1] = *(const v8bf*)(vp + 16);
      O[tt] = wmma_bf16(PA.v, VB.v, O[tt]);
    }
  }

  // --- epilogue: normalize, residual, write both tuple outputs ---
  float g = gamma[0];
  const size_t xbase = (size_t)b * C_ * N_;
#pragma unroll
  for (int tt = 0; tt < 16; tt++) {
#pragma unroll
    for (int r = 0; r < 8; r++) {
      int n = n0 + r + hs * 8;
      int c = tt * 16 + rA;
      size_t idx = xbase + (size_t)c * N_ + n;
      float y = x[idx] + g * (O[tt][r] / lrow[r]);
      out[idx] = y;
      out[idx + (size_t)B_ * C_ * N_] = y;
    }
  }
}

// ---------------------------------------------------------------------------
extern "C" void kernel_launch(void* const* d_in, const int* in_sizes, int n_in,
                              void* d_out, int out_size, void* d_ws, size_t ws_size,
                              hipStream_t stream) {
  const float* x     = (const float*)d_in[0];
  // d_in[1] = temb (unused by the reference math)
  const float* Wq    = (const float*)d_in[2];
  const float* bq    = (const float*)d_in[3];
  const float* Wk    = (const float*)d_in[4];
  const float* bk    = (const float*)d_in[5];
  const float* Wv    = (const float*)d_in[6];
  const float* bv    = (const float*)d_in[7];
  const float* gamma = (const float*)d_in[8];
  float* out = (float*)d_out;

  char* ws = (char*)d_ws;
  constexpr size_t OFF_XT = 0;                       // 8,388,608  bf16 [B][N][C]
  constexpr size_t OFF_VT = 8388608;                 // 8,388,608  bf16 [B][C][N]
  constexpr size_t OFF_QB = 16777216;                // 1,048,576  bf16 [B][N][32]
  constexpr size_t OFF_KB = 17825792;                // 1,048,576  bf16 [B][N][32]
  constexpr size_t OFF_WQ = 18874368;                // 16,384
  constexpr size_t OFF_WK = 18890752;                // 16,384
  constexpr size_t OFF_WV = 18907136;                // 131,072
  bf16_t* Xt  = (bf16_t*)(ws + OFF_XT);
  bf16_t* Vt  = (bf16_t*)(ws + OFF_VT);
  bf16_t* Qbw = (bf16_t*)(ws + OFF_QB);
  bf16_t* Kbw = (bf16_t*)(ws + OFF_KB);
  bf16_t* Wqb = (bf16_t*)(ws + OFF_WQ);
  bf16_t* Wkb = (bf16_t*)(ws + OFF_WK);
  bf16_t* Wvb = (bf16_t*)(ws + OFF_WV);

  convert_x_kernel<<<(B_ * C_ * N_) / 256, 256, 0, stream>>>(x, Xt);
  convert_w_kernel<<<320, 256, 0, stream>>>(Wq, Wk, Wv, Wqb, Wkb, Wvb);

  proj_kernel<<<2560, 256, 0, stream>>>(Xt, Wqb, Wkb, Wvb, bq, bk, bv,
                                        Qbw, Kbw, Vt);

  attn_kernel<<<256, 128, 0, stream>>>(Qbw, Kbw, Vt, x, gamma, out);
}
